// attn_block_86655260164950
// MI455X (gfx1250) — matrix operand order repaired and checked by reading, not compiled, not run on hardware
//
#include <hip/hip_runtime.h>
#include <hip/hip_bf16.h>
#include <stdint.h>

// ---------------- problem constants ----------------
#define BB   8
#define NN   4096
#define MM   256
#define CC   512
#define HH   8
#define KK   16
#define HID  2048
#define LL   (NN + MM)          // 4352
#define DH   (CC / HH)          // 64
#define MT   (BB * LL)          // 34816 rows total (multiple of 128)

typedef __attribute__((ext_vector_type(16))) _Float16 v16h;
typedef __attribute__((ext_vector_type(8)))  _Float16 v8h;
typedef __attribute__((ext_vector_type(8)))  float    v8f;
typedef __attribute__((ext_vector_type(4)))  unsigned int v4u;
typedef __attribute__((ext_vector_type(8)))  int v8i;
typedef __attribute__((ext_vector_type(4)))  int v4i;

union F16Frag { v16h v; v8h h[2]; };

__device__ __forceinline__ v8f wmma_f16(v16h a, v16h b, v8f c) {
    // D = A(16x32 f16) * B(32x16 f16) + C(16x16 f32)
    return __builtin_amdgcn_wmma_f32_16x16x32_f16(
        /*neg_a=*/false, a, /*neg_b=*/false, b,
        /*c_mod=*/(short)0, c, /*reuse_a=*/false, /*reuse_b=*/false);
}

// ---------------- TDM: DMA a 64-row x 128-half weight tile into LDS ----------------
// Rows are padded by the TDM pad feature: after every 64 DWORDs (one 256B row)
// insert 4 DWORDs (16B) -> LDS row stride = 272B = 136 halves (bank-conflict free).
#define TILE_KH   128            // halves per row per tile
#define TILE_NR   64             // weight rows per tile
#define LDS_ROWH  136            // padded row stride in halves
#define LDS_TILEH (TILE_NR * LDS_ROWH)

__device__ __forceinline__ void tdm_load_b_tile(const _Float16* gsrc, uint32_t lds_off,
                                                int Kt, int Nt) {
    v4u g0; v8i g1; v4i z4 = {0, 0, 0, 0};
    uint64_t ga = (uint64_t)(uintptr_t)gsrc;
    g0[0] = 1u;                                   // count=1, user descriptor
    g0[1] = lds_off;                              // LDS byte address
    g0[2] = (uint32_t)ga;                         // global_addr[31:0]
    g0[3] = (uint32_t)((ga >> 32) & 0x01FFFFFFu)  // global_addr[56:32]
          | 0x80000000u;                          // type=2 ("image")
    // data_size=1 (2B), pad_enable=1, pad_interval=5 (64 DW), pad_amount=3 (4 DW)
    g1[0] = (1 << 16) | (1 << 20) | (5 << 22) | (3 << 25);
    g1[1] = (int)(((uint32_t)Kt & 0xFFFFu) << 16);                 // tensor_dim0 lo16
    g1[2] = (int)((((uint32_t)Kt >> 16) & 0xFFFFu)
          | (((uint32_t)Nt & 0xFFFFu) << 16));                     // dim0 hi | dim1 lo
    g1[3] = (int)((((uint32_t)Nt >> 16) & 0xFFFFu)
          | ((uint32_t)TILE_KH << 16));                            // dim1 hi | tile_dim0
    g1[4] = TILE_NR;                                               // tile_dim1 | tile_dim2=0
    g1[5] = Kt;                                                    // tensor_dim0_stride lo32
    g1[6] = 0;                                                     // stride0 hi | stride1 lo
    g1[7] = 0;
#if __has_include(<hip/amd_detail/amd_gfx1250_TDM.h>)
    v8i z8 = {0, 0, 0, 0, 0, 0, 0, 0};
    __builtin_amdgcn_tensor_load_to_lds(g0, g1, z4, z4, z8, 0);
#else
    __builtin_amdgcn_tensor_load_to_lds(g0, g1, z4, z4, 0);
#endif
}

// ---------------- f32 -> f16 convert ----------------
__global__ void cvt_f16_kernel(const float* __restrict__ s, _Float16* __restrict__ d, int n) {
    int i = blockIdx.x * 256 + threadIdx.x;
    if (i < n) d[i] = (_Float16)s[i];
}

// ---------------- LayerNorm (optionally fused concat of x / seman_x) ----------------
__global__ void ln_kernel(const float* __restrict__ x, const float* __restrict__ sx,
                          const float* __restrict__ g, const float* __restrict__ beta,
                          _Float16* __restrict__ out) {
    int row = blockIdx.x;
    const float* src;
    if (sx) {
        int b = row / LL, l = row % LL;
        src = (l < NN) ? (x + ((size_t)b * NN + l) * CC)
                       : (sx + ((size_t)b * MM + (l - NN)) * CC);
    } else {
        src = x + (size_t)row * CC;
    }
    int tid = threadIdx.x;
    float v0 = src[2 * tid], v1 = src[2 * tid + 1];
    float s = v0 + v1, s2 = v0 * v0 + v1 * v1;
    #pragma unroll
    for (int o = 16; o > 0; o >>= 1) { s += __shfl_xor(s, o); s2 += __shfl_xor(s2, o); }
    __shared__ float ws[8], ws2[8];
    int wv = tid >> 5, ln = tid & 31;
    if (ln == 0) { ws[wv] = s; ws2[wv] = s2; }
    __syncthreads();
    if (tid == 0) {
        float a = 0.f, b2 = 0.f;
        #pragma unroll
        for (int i = 0; i < 8; ++i) { a += ws[i]; b2 += ws2[i]; }
        ws[0] = a; ws2[0] = b2;
    }
    __syncthreads();
    float mean = ws[0] * (1.0f / CC);
    float var  = ws2[0] * (1.0f / CC) - mean * mean;
    float inv  = rsqrtf(var + 1e-5f);
    int c0 = 2 * tid;
    out[(size_t)row * CC + c0]     = (_Float16)((v0 - mean) * inv * g[c0]     + beta[c0]);
    out[(size_t)row * CC + c0 + 1] = (_Float16)((v1 - mean) * inv * g[c0 + 1] + beta[c0 + 1]);
}

// ---------------- KNN top-16 by squared 2D distance ----------------
__global__ void topk_kernel(const float* __restrict__ coords, const float* __restrict__ scoords,
                            const unsigned char* __restrict__ mask, int* __restrict__ idxout) {
    __shared__ float dist[LL];
    __shared__ float rv[256];
    __shared__ int   ri[256];
    int row = blockIdx.x;
    int b = row / LL, l = row % LL;
    int tid = threadIdx.x;
    float qx, qy;
    {
        const float* qc = (l < NN) ? (coords + ((size_t)b * NN + l) * 2)
                                   : (scoords + ((size_t)b * MM + (l - NN)) * 2);
        qx = qc[0]; qy = qc[1];
    }
    for (int j = tid; j < LL; j += 256) {
        const float* cc = (j < NN) ? (coords + ((size_t)b * NN + j) * 2)
                                   : (scoords + ((size_t)b * MM + (j - NN)) * 2);
        float dx = qx - cc[0], dy = qy - cc[1];
        float d = dx * dx + dy * dy;
        if (!mask[(size_t)b * LL + j]) d = 3.0e38f;
        dist[j] = d;
    }
    __syncthreads();
    for (int it = 0; it < KK; ++it) {
        float best = 3.0e38f; int bi = 0;
        for (int j = tid; j < LL; j += 256) {
            float d = dist[j];
            if (d < best) { best = d; bi = j; }
        }
        rv[tid] = best; ri[tid] = bi;
        __syncthreads();
        for (int sft = 128; sft > 0; sft >>= 1) {
            if (tid < sft && rv[tid + sft] < rv[tid]) { rv[tid] = rv[tid + sft]; ri[tid] = ri[tid + sft]; }
            __syncthreads();
        }
        if (tid == 0) { idxout[(size_t)row * KK + it] = ri[0]; dist[ri[0]] = 3.0e38f; }
        __syncthreads();
    }
}

// ---------------- gather + softmax(K=16) + weighted sum ----------------
__global__ void attn_kernel(const _Float16* __restrict__ qkv, const int* __restrict__ idx,
                            const unsigned char* __restrict__ mask, _Float16* __restrict__ aout) {
    int row = blockIdx.x;
    int b = row / LL;
    int tid = threadIdx.x;
    int h = tid >> 5, lane = tid & 31;
    const _Float16* qb = qkv + (size_t)row * (3 * CC) + h * DH;
    float q0 = (float)qb[2 * lane], q1 = (float)qb[2 * lane + 1];
    float logit[KK];
    int   nbr[KK];
    const float scale = 0.125f;   // DH^-0.5
    #pragma unroll
    for (int k = 0; k < KK; ++k) {
        int j = idx[(size_t)row * KK + k];
        nbr[k] = j;
        const _Float16* kb = qkv + ((size_t)b * LL + j) * (3 * CC) + CC + h * DH;
        float p = q0 * (float)kb[2 * lane] + q1 * (float)kb[2 * lane + 1];
        #pragma unroll
        for (int o = 16; o > 0; o >>= 1) p += __shfl_xor(p, o);
        p *= scale;
        if (!mask[(size_t)b * LL + j]) p = -1.0e9f;
        logit[k] = p;
    }
    float mx = -3.0e38f;
    #pragma unroll
    for (int k = 0; k < KK; ++k) mx = fmaxf(mx, logit[k]);
    float sum = 0.f;
    #pragma unroll
    for (int k = 0; k < KK; ++k) { logit[k] = __expf(logit[k] - mx); sum += logit[k]; }
    float inv = 1.0f / sum;
    float o0 = 0.f, o1 = 0.f;
    #pragma unroll
    for (int k = 0; k < KK; ++k) {
        const _Float16* vb = qkv + ((size_t)b * LL + nbr[k]) * (3 * CC) + 2 * CC + h * DH;
        float a = logit[k] * inv;
        o0 += a * (float)vb[2 * lane];
        o1 += a * (float)vb[2 * lane + 1];
    }
    aout[(size_t)row * CC + h * DH + 2 * lane]     = (_Float16)o0;
    aout[(size_t)row * CC + h * DH + 2 * lane + 1] = (_Float16)o1;
}

// ---------------- WMMA GEMM with TDM-staged weight tiles ----------------
// D(Mt x Nt) = A(Mt x Kt) * W(Nt x Kt)^T + epilogue.
// Block: 256 threads = 8 waves -> 128 rows x 64 cols; the 64x128h weight tile is
// DMA'd to LDS (TENSOR_LOAD_TO_LDS, double-buffered) and shared by all 8 waves.
// Inner chunk loop is software-pipelined; __builtin_amdgcn_sched_barrier(0) pins
// the step-(i+1) loads above the step-i WMMAs so LDS/VMEM latency is hidden.
__device__ __forceinline__ void load_afrag(const _Float16* __restrict__ arow, int kk, F16Frag& a) {
    a.h[0] = *(const v8h*)(arow + kk);
    a.h[1] = *(const v8h*)(arow + kk + 8);
}

__device__ __forceinline__ void load_bfrags(const _Float16* __restrict__ lb, int ks,
                                            int lb0, int lb1, int lb2, int lb3,
                                            F16Frag& b0, F16Frag& b1, F16Frag& b2, F16Frag& b3) {
    b0.h[0] = *(const v8h*)(lb + lb0 + ks); b0.h[1] = *(const v8h*)(lb + lb0 + ks + 8);
    b1.h[0] = *(const v8h*)(lb + lb1 + ks); b1.h[1] = *(const v8h*)(lb + lb1 + ks + 8);
    b2.h[0] = *(const v8h*)(lb + lb2 + ks); b2.h[1] = *(const v8h*)(lb + lb2 + ks + 8);
    b3.h[0] = *(const v8h*)(lb + lb3 + ks); b3.h[1] = *(const v8h*)(lb + lb3 + ks + 8);
}

__device__ __forceinline__ void epilogue_tile(
    v8f acc, int n, int mbase, int Nt, int mode,
    const float* __restrict__ bias, const float* __restrict__ ls,
    const float* __restrict__ resid, const float* __restrict__ rx, const float* __restrict__ rsem,
    float* __restrict__ outF, _Float16* __restrict__ outH) {
    float bv = bias ? bias[n] : 0.0f;
    float lv = ls ? ls[n] : 1.0f;
    #pragma unroll
    for (int r = 0; r < 8; ++r) {
        int m = mbase + r;
        float v = acc[r] + bv;
        if (mode == 0) {
            outH[(size_t)m * Nt + n] = (_Float16)v;
        } else if (mode == 1) {
            int b = m / LL, l = m % LL;
            float res = (l < NN) ? rx[((size_t)b * NN + l) * CC + n]
                                 : rsem[((size_t)b * MM + (l - NN)) * CC + n];
            outF[(size_t)m * Nt + n] = res + lv * v;
        } else if (mode == 2) {
            float t = v;
            float gg = 0.5f * t * (1.0f + tanhf(0.7978845608f * (t + 0.044715f * t * t * t)));
            outH[(size_t)m * Nt + n] = (_Float16)gg;
        } else {
            float res = resid[(size_t)m * Nt + n];
            outF[(size_t)m * Nt + n] = res + lv * v;
        }
    }
}

__global__ __launch_bounds__(256) void wmma_gemm_kernel(
    const _Float16* __restrict__ A, const _Float16* __restrict__ W,
    int Mt, int Nt, int Kt, int mode,
    const float* __restrict__ bias, const float* __restrict__ ls,
    const float* __restrict__ resid, const float* __restrict__ rx, const float* __restrict__ rsem,
    float* __restrict__ outF, _Float16* __restrict__ outH) {
    __shared__ _Float16 ldsb[2 * LDS_TILEH];   // double-buffered padded weight tiles
    int lane = threadIdx.x & 31;
    int wave = threadIdx.x >> 5;
    int m0 = (blockIdx.y * 8 + wave) << 4;   // 16 rows per wave
    int n0 = blockIdx.x << 6;                // 64 cols per block
    int lr = lane & 15, lh = lane >> 4;

    // A fragment source (contiguous k order, same as B): lane holds row m0+lr,
    // 16 contiguous halves [k..k+15]; upper half-wave offset +16.
    const _Float16* arow = A + (size_t)(m0 + lr) * Kt + (lh << 4);
    // LDS B fragment bases (halves): lane lr owns weight row 16*t+lr,
    // 16 contiguous halves at local k (upper half-wave +16).
    int lb0 = (( 0 + lr) * LDS_ROWH) + (lh << 4);
    int lb1 = ((16 + lr) * LDS_ROWH) + (lh << 4);
    int lb2 = ((32 + lr) * LDS_ROWH) + (lh << 4);
    int lb3 = ((48 + lr) * LDS_ROWH) + (lh << 4);

    const _Float16* wtile = W + (size_t)n0 * Kt;       // block-uniform
    uint32_t lds_base = (uint32_t)(uintptr_t)(&ldsb[0]);

    int nchunks = Kt >> 7;                             // 128 halves per chunk
    if (wave == 0) tdm_load_b_tile(wtile, lds_base, Kt, Nt);

    v8f acc0 = {}, acc1 = {}, acc2 = {}, acc3 = {};
    for (int c = 0; c < nchunks; ++c) {
        if (wave == 0) {
            if (c + 1 < nchunks) {
                tdm_load_b_tile(wtile + ((c + 1) << 7),
                                lds_base + (uint32_t)(((c + 1) & 1) * LDS_TILEH * 2),
                                Kt, Nt);
                __builtin_amdgcn_s_wait_tensorcnt(1);  // chunk c resident (in-order TDM)
            } else {
                __builtin_amdgcn_s_wait_tensorcnt(0);
            }
        }
        __syncthreads();
        const _Float16* lb = ldsb + (c & 1) * LDS_TILEH;
        int kk0 = c << 7;

        // software-pipelined chunk: preload step 0; inside the loop, loads for
        // step i+1 are pinned ABOVE the WMMAs of step i by a sched_barrier, so
        // they stay in flight while the matrix pipe works.
        F16Frag a, b0, b1, b2, b3;
        load_afrag(arow, kk0, a);
        load_bfrags(lb, 0, lb0, lb1, lb2, lb3, b0, b1, b2, b3);
        #pragma unroll
        for (int ks = 0; ks < TILE_KH; ks += 32) {
            F16Frag an, b0n, b1n, b2n, b3n;
            if (ks + 32 < TILE_KH) {
                load_afrag(arow, kk0 + ks + 32, an);
                load_bfrags(lb, ks + 32, lb0, lb1, lb2, lb3, b0n, b1n, b2n, b3n);
                __builtin_prefetch(arow + kk0 + ks + TILE_KH, 0, 1);  // global_prefetch_b8
            }
            __builtin_amdgcn_sched_barrier(0);   // loads above stay above the WMMAs
            acc0 = wmma_f16(a.v, b0.v, acc0);
            acc1 = wmma_f16(a.v, b1.v, acc1);
            acc2 = wmma_f16(a.v, b2.v, acc2);
            acc3 = wmma_f16(a.v, b3.v, acc3);
            __builtin_amdgcn_sched_barrier(0);   // keep next-step loads out of this group
            if (ks + 32 < TILE_KH) { a = an; b0 = b0n; b1 = b1n; b2 = b2n; b3 = b3n; }
        }
        __syncthreads();   // all waves done with buf(c&1) before it is overwritten
    }
    int mbase = m0 + (lh << 3);
    epilogue_tile(acc0, n0 +  0 + lr, mbase, Nt, mode, bias, ls, resid, rx, rsem, outF, outH);
    epilogue_tile(acc1, n0 + 16 + lr, mbase, Nt, mode, bias, ls, resid, rx, rsem, outF, outH);
    epilogue_tile(acc2, n0 + 32 + lr, mbase, Nt, mode, bias, ls, resid, rx, rsem, outF, outH);
    epilogue_tile(acc3, n0 + 48 + lr, mbase, Nt, mode, bias, ls, resid, rx, rsem, outF, outH);
}

// ---------------- host side ----------------
extern "C" void kernel_launch(void* const* d_in, const int* in_sizes, int n_in,
                              void* d_out, int out_size, void* d_ws, size_t ws_size,
                              hipStream_t stream) {
    const float* x       = (const float*)d_in[0];
    const float* sx      = (const float*)d_in[1];
    const float* coords  = (const float*)d_in[2];
    const float* scoords = (const float*)d_in[3];
    const unsigned char* mask = (const unsigned char*)d_in[4];
    const float* n1_g   = (const float*)d_in[5];
    const float* n1_b   = (const float*)d_in[6];
    const float* qkv_w  = (const float*)d_in[7];
    const float* qkv_b  = (const float*)d_in[8];
    const float* proj_w = (const float*)d_in[9];
    const float* proj_b = (const float*)d_in[10];
    const float* ls1    = (const float*)d_in[11];
    const float* n2_g   = (const float*)d_in[12];
    const float* n2_b   = (const float*)d_in[13];
    const float* fc1_w  = (const float*)d_in[14];
    const float* fc1_b  = (const float*)d_in[15];
    const float* fc2_w  = (const float*)d_in[16];
    const float* fc2_b  = (const float*)d_in[17];
    const float* ls2    = (const float*)d_in[18];
    float* out = (float*)d_out;

    char* ws = (char*)d_ws;
    size_t off = 0;
    auto alloc = [&](size_t bytes) -> char* {
        char* p = ws + off;
        off = (off + bytes + 255) & ~(size_t)255;
        return p;
    };

    _Float16* wq16  = (_Float16*)alloc((size_t)3 * CC * CC * sizeof(_Float16));
    _Float16* wp16  = (_Float16*)alloc((size_t)CC * CC * sizeof(_Float16));
    _Float16* wf116 = (_Float16*)alloc((size_t)HID * CC * sizeof(_Float16));
    _Float16* wf216 = (_Float16*)alloc((size_t)CC * HID * sizeof(_Float16));
    _Float16* h16   = (_Float16*)alloc((size_t)MT * CC * sizeof(_Float16));
    _Float16* qkv16 = (_Float16*)alloc((size_t)MT * 3 * CC * sizeof(_Float16));
    int*      knn   = (int*)     alloc((size_t)MT * KK * sizeof(int));
    _Float16* att16 = (_Float16*)alloc((size_t)MT * CC * sizeof(_Float16));
    float*    xr32  = (float*)   alloc((size_t)MT * CC * sizeof(float));
    _Float16* h2_16 = (_Float16*)alloc((size_t)MT * CC * sizeof(_Float16));
    _Float16* a1_16 = (_Float16*)alloc((size_t)MT * HID * sizeof(_Float16));

    // weight converts (f32 -> f16)
    {
        int n;
        n = 3 * CC * CC; cvt_f16_kernel<<<(n + 255) / 256, 256, 0, stream>>>(qkv_w, wq16, n);
        n = CC * CC;     cvt_f16_kernel<<<(n + 255) / 256, 256, 0, stream>>>(proj_w, wp16, n);
        n = HID * CC;    cvt_f16_kernel<<<(n + 255) / 256, 256, 0, stream>>>(fc1_w, wf116, n);
        n = CC * HID;    cvt_f16_kernel<<<(n + 255) / 256, 256, 0, stream>>>(fc2_w, wf216, n);
    }

    // LN1 fused with concat(x, seman_x)
    ln_kernel<<<MT, 256, 0, stream>>>(x, sx, n1_g, n1_b, h16);

    // KNN top-16 on 2D coords
    topk_kernel<<<MT, 256, 0, stream>>>(coords, scoords, mask, knn);

    // QKV = h @ qkv_w^T + qkv_b   (f16 out)
    wmma_gemm_kernel<<<dim3(3 * CC / 64, MT / 128), 256, 0, stream>>>(
        h16, wq16, MT, 3 * CC, CC, /*mode=*/0,
        qkv_b, nullptr, nullptr, nullptr, nullptr, nullptr, qkv16);

    // gather K/V of 16 neighbors, softmax, weighted sum
    attn_kernel<<<MT, 256, 0, stream>>>(qkv16, knn, mask, att16);

    // xr = concat(x,seman_x) + ls1 * (att @ proj_w^T + proj_b)   (f32 out)
    wmma_gemm_kernel<<<dim3(CC / 64, MT / 128), 256, 0, stream>>>(
        att16, wp16, MT, CC, CC, /*mode=*/1,
        proj_b, ls1, nullptr, x, sx, xr32, nullptr);

    // LN2
    ln_kernel<<<MT, 256, 0, stream>>>(xr32, nullptr, n2_g, n2_b, h2_16);

    // a1 = gelu(h2 @ fc1_w^T + fc1_b)   (f16 out)
    wmma_gemm_kernel<<<dim3(HID / 64, MT / 128), 256, 0, stream>>>(
        h2_16, wf116, MT, HID, CC, /*mode=*/2,
        fc1_b, nullptr, nullptr, nullptr, nullptr, nullptr, a1_16);

    // out = xr + ls2 * (a1 @ fc2_w^T + fc2_b)   (f32 out -> d_out)
    wmma_gemm_kernel<<<dim3(CC / 64, MT / 128), 256, 0, stream>>>(
        a1_16, wf216, MT, CC, HID, /*mode=*/3,
        fc2_b, ls2, xr32, nullptr, nullptr, out, nullptr);
}
